// GTLayer_11905649344608
// MI455X (gfx1250) — matrix operand order repaired and checked by reading, not compiled, not run on hardware
//
#include <hip/hip_runtime.h>
#include <hip/hip_bf16.h>

typedef __attribute__((ext_vector_type(16))) __bf16 v16bf;
typedef __attribute__((ext_vector_type(8)))  __bf16 v8bf;
typedef __attribute__((ext_vector_type(8)))  float  v8f;

#define NNODES   4096
#define HIDDEN   256
#define HEADS    8
#define HEAD_DIM 32
#define EPS      1e-5f

static __device__ __forceinline__ v16bf cat16(v8bf a, v8bf b) {
  return __builtin_shufflevector(a, b, 0,1,2,3,4,5,6,7,8,9,10,11,12,13,14,15);
}

// ---- CDNA5 async global->LDS copy (16B per lane), with safe fallback --------
#if __has_builtin(__builtin_amdgcn_global_load_async_to_lds_b128)
#define HAVE_ASYNC_LDS 1
typedef int v4i_async __attribute__((ext_vector_type(4)));
typedef __attribute__((address_space(1))) v4i_async gas_v4i;  // global (AS1)
typedef __attribute__((address_space(3))) v4i_async las_v4i;  // LDS (AS3)
#endif

static __device__ __forceinline__ void async_copy16(const void* g, void* l) {
#ifdef HAVE_ASYNC_LDS
  __builtin_amdgcn_global_load_async_to_lds_b128((gas_v4i*)g, (las_v4i*)l, 0, 0);
#else
  *(v8bf*)l = *(const v8bf*)g;
#endif
}
static __device__ __forceinline__ void async_wait() {
#ifdef HAVE_ASYNC_LDS
  asm volatile("s_wait_asynccnt 0x0" ::: "memory");
#endif
}

// ---------------------------------------------------------------------------
// WMMA GEMM: out = op(X[M,K] @ W[N,K]^T + bias)*scale (+ residual)
// 256 threads = 8 waves; block tile 128(M) x 32(N); K step 32.
// Wave w owns rows w*16..w*16+15 and computes both 16-wide N tiles,
// reusing one A fragment for two WMMAs.
// ---------------------------------------------------------------------------
template<bool RELU, bool BF16OUT>
__global__ __launch_bounds__(256)
void gemm_bias_kernel(const float* __restrict__ X, const float* __restrict__ W,
                      const float* __restrict__ bias, const float* __restrict__ residual,
                      float* __restrict__ outF, __bf16* __restrict__ outB,
                      int M, int N, int K, float scale)
{
  __shared__ __align__(16) __bf16 Xs[128][40];  // 128 rows x 32 k (+8 pad)
  __shared__ __align__(16) __bf16 Wsh[32][40];  // 32 n-rows x 32 k (+8 pad)

  const int tid  = threadIdx.x;
  const int m0   = blockIdx.x * 128;
  const int n0   = blockIdx.y * 32;
  const int lane = tid & 31;
  const int wave = tid >> 5;
  const int lm   = lane & 15;
  const int half = lane >> 4;

  v8f acc0 = {}, acc1 = {};
  for (int k0 = 0; k0 < K; k0 += 32) {
    // Stage X tile (128x32 f32 -> bf16): 1024 float4 / 256 threads = 4 each
    #pragma unroll
    for (int it = 0; it < 4; ++it) {
      const int idx = tid + it * 256;
      const int r = idx >> 3, c4 = (idx & 7) * 4;
      const float4 xv = *(const float4*)(X + (size_t)(m0 + r) * K + k0 + c4);
      Xs[r][c4 + 0] = (__bf16)xv.x; Xs[r][c4 + 1] = (__bf16)xv.y;
      Xs[r][c4 + 2] = (__bf16)xv.z; Xs[r][c4 + 3] = (__bf16)xv.w;
    }
    // Stage W tile (32 n-rows x 32 k): one float4 per thread
    {
      const int r = tid >> 3, c4 = (tid & 7) * 4;
      const float4 wv = *(const float4*)(W + (size_t)(n0 + r) * K + k0 + c4);
      Wsh[r][c4 + 0] = (__bf16)wv.x; Wsh[r][c4 + 1] = (__bf16)wv.y;
      Wsh[r][c4 + 2] = (__bf16)wv.z; Wsh[r][c4 + 3] = (__bf16)wv.w;
    }
    __syncthreads();

    // A fragment (16x32): lane m=lm; elem e -> k=half*16+e (contiguous)
    v8bf a0 = *(const v8bf*)&Xs[wave * 16 + lm][half * 16];
    v8bf a1 = *(const v8bf*)&Xs[wave * 16 + lm][half * 16 + 8];
    const v16bf a = cat16(a0, a1);
    // B fragments (32x16): lane n=lm; elem e -> k=half*16+e (contiguous)
    v8bf b00 = *(const v8bf*)&Wsh[lm][half * 16];
    v8bf b01 = *(const v8bf*)&Wsh[lm][half * 16 + 8];
    v8bf b10 = *(const v8bf*)&Wsh[16 + lm][half * 16];
    v8bf b11 = *(const v8bf*)&Wsh[16 + lm][half * 16 + 8];
    acc0 = __builtin_amdgcn_wmma_f32_16x16x32_bf16(false, a, false, cat16(b00, b01),
                                                   (short)0, acc0, false, false);
    acc1 = __builtin_amdgcn_wmma_f32_16x16x32_bf16(false, a, false, cat16(b10, b11),
                                                   (short)0, acc1, false, false);
    __syncthreads();
  }

  // Epilogue: C layout — lane column n = lm; VGPR r holds row half*8+r
  const int na = n0 + lm;
  const int nb = n0 + 16 + lm;
  const float bva = bias[na];
  const float bvb = bias[nb];
  #pragma unroll
  for (int r = 0; r < 8; ++r) {
    const int m = m0 + wave * 16 + half * 8 + r;
    float va = (acc0[r] + bva) * scale;
    float vb = (acc1[r] + bvb) * scale;
    if (RELU) { va = fmaxf(va, 0.0f); vb = fmaxf(vb, 0.0f); }
    if (residual) {
      va += residual[(size_t)m * N + na];
      vb += residual[(size_t)m * N + nb];
    }
    if (BF16OUT) {
      outB[(size_t)m * N + na] = (__bf16)va;
      outB[(size_t)m * N + nb] = (__bf16)vb;
    } else {
      outF[(size_t)m * N + na] = va;
      outF[(size_t)m * N + nb] = vb;
    }
  }
}

// ---------------------------------------------------------------------------
// Fused masked attention (flash-style online softmax).
// 128 threads = 4 waves; each wave owns 16 query rows; 64-key chunks
// (4 S tiles) so one cross-lane max/sum tree covers 64 keys.
// logits = (Q K^T) * Amask ; softmax over keys ; O = P V.
// ---------------------------------------------------------------------------
__global__ __launch_bounds__(128)
void attn_kernel(const __bf16* __restrict__ Q, const __bf16* __restrict__ Km,
                 const __bf16* __restrict__ Vm, const float* __restrict__ Amask,
                 float* __restrict__ Out)
{
  __shared__ __align__(16) __bf16 Ks[64][40];      // [key][k]       (async staged)
  __shared__ __align__(16) __bf16 Vt[32][72];      // [d][key]       (transposed)
  __shared__ __align__(16) __bf16 Ps[4][16][72];   // per-wave P [row][key]

  const int tid  = threadIdx.x;
  const int head = blockIdx.x >> 6;
  const int rb   = (blockIdx.x & 63) * 64;
  const int lane = tid & 31;
  const int wave = tid >> 5;
  const int lm   = lane & 15;
  const int half = lane >> 4;
  const size_t hbase = (size_t)head * (NNODES * HEAD_DIM);

  // Q A-fragment for this wave's 16 rows (registers, loop-invariant)
  const int qrow = rb + wave * 16 + lm;
  v8bf qa0 = *(const v8bf*)(Q + hbase + (size_t)qrow * 32 + half * 16);
  v8bf qa1 = *(const v8bf*)(Q + hbase + (size_t)qrow * 32 + half * 16 + 8);
  const v16bf qa = cat16(qa0, qa1);

  float mrow[8], lrow[8];
  #pragma unroll
  for (int r = 0; r < 8; ++r) { mrow[r] = -1e30f; lrow[r] = 0.0f; }
  v8f o0 = {}, o1 = {};
  const int myrow0 = rb + wave * 16 + half * 8;

  for (int j0 = 0; j0 < NNODES; j0 += 64) {
    // Stage K (async DMA to LDS) and V (manual transpose): 64 keys x 32 dims
    #pragma unroll
    for (int it = 0; it < 2; ++it) {
      const int c = tid + it * 128;        // 256 chunks of 16B
      const int key = c >> 2;
      const int dg  = (c & 3) * 8;
      async_copy16(Km + hbase + (size_t)(j0 + key) * 32 + dg, &Ks[key][dg]);
      v8bf vv = *(const v8bf*)(Vm + hbase + (size_t)(j0 + key) * 32 + dg);
      #pragma unroll
      for (int e = 0; e < 8; ++e) Vt[dg + e][key] = vv[e];
    }
    async_wait();
    __syncthreads();

    // S tiles: Q (16x32) @ K^T -> four 16x16 tiles (keys j0+16t .. +15)
    v8f s[4];
    #pragma unroll
    for (int t = 0; t < 4; ++t) {
      v8bf kb0 = *(const v8bf*)&Ks[t * 16 + lm][half * 16];
      v8bf kb1 = *(const v8bf*)&Ks[t * 16 + lm][half * 16 + 8];
      v8f z = {};
      s[t] = __builtin_amdgcn_wmma_f32_16x16x32_bf16(false, qa, false, cat16(kb0, kb1),
                                                     (short)0, z, false, false);
    }

    // Mask + online softmax; one reduction tree per row covers all 64 keys
    #pragma unroll
    for (int r = 0; r < 8; ++r) {
      const size_t arow = (size_t)(myrow0 + r) * NNODES;
      float tv[4];
      #pragma unroll
      for (int t = 0; t < 4; ++t)
        tv[t] = s[t][r] * Amask[arow + j0 + t * 16 + lm];

      float mx = fmaxf(fmaxf(tv[0], tv[1]), fmaxf(tv[2], tv[3]));
      #pragma unroll
      for (int xm = 1; xm < 16; xm <<= 1)
        mx = fmaxf(mx, __shfl_xor(mx, xm, 32));
      const float mnew = fmaxf(mrow[r], mx);
      const float fsc  = __expf(mrow[r] - mnew);

      float p[4];
      #pragma unroll
      for (int t = 0; t < 4; ++t) p[t] = __expf(tv[t] - mnew);
      float ps = (p[0] + p[1]) + (p[2] + p[3]);
      #pragma unroll
      for (int xm = 1; xm < 16; xm <<= 1)
        ps += __shfl_xor(ps, xm, 32);

      lrow[r] = lrow[r] * fsc + ps;
      mrow[r] = mnew;
      o0[r] *= fsc;
      o1[r] *= fsc;
      #pragma unroll
      for (int t = 0; t < 4; ++t)
        Ps[wave][half * 8 + r][t * 16 + lm] = (__bf16)p[t];
    }

    // O += P (16x64) @ V (64x32): K split 2 x 32, N split 2 x 16
    v8bf pk00 = *(const v8bf*)&Ps[wave][lm][half * 16];
    v8bf pk01 = *(const v8bf*)&Ps[wave][lm][half * 16 + 8];
    v8bf pk10 = *(const v8bf*)&Ps[wave][lm][32 + half * 16];
    v8bf pk11 = *(const v8bf*)&Ps[wave][lm][32 + half * 16 + 8];
    const v16bf pa0 = cat16(pk00, pk01);   // keys 0..31
    const v16bf pa1 = cat16(pk10, pk11);   // keys 32..63
    #pragma unroll
    for (int kk = 0; kk < 2; ++kk) {
      const v16bf pa = (kk == 0) ? pa0 : pa1;
      v8bf vb00 = *(const v8bf*)&Vt[lm][kk * 32 + half * 16];
      v8bf vb01 = *(const v8bf*)&Vt[lm][kk * 32 + half * 16 + 8];
      v8bf vb10 = *(const v8bf*)&Vt[16 + lm][kk * 32 + half * 16];
      v8bf vb11 = *(const v8bf*)&Vt[16 + lm][kk * 32 + half * 16 + 8];
      o0 = __builtin_amdgcn_wmma_f32_16x16x32_bf16(false, pa, false, cat16(vb00, vb01),
                                                   (short)0, o0, false, false);
      o1 = __builtin_amdgcn_wmma_f32_16x16x32_bf16(false, pa, false, cat16(vb10, vb11),
                                                   (short)0, o1, false, false);
    }
    __syncthreads();
  }

  // Normalize and write per-head output (flat layout == raw reshape back)
  #pragma unroll
  for (int r = 0; r < 8; ++r) {
    const float inv = 1.0f / lrow[r];
    const size_t row = (size_t)(myrow0 + r);
    Out[hbase + row * 32 + lm]      = o0[r] * inv;
    Out[hbase + row * 32 + 16 + lm] = o1[r] * inv;
  }
}

// ---------------------------------------------------------------------------
// BatchNorm1d (training stats over the node axis)
// ---------------------------------------------------------------------------
__global__ __launch_bounds__(256)
void bn_stats_kernel(const float* __restrict__ X, float* __restrict__ mean,
                     float* __restrict__ rstd)
{
  __shared__ float sh[256], sh2[256];
  const int j = blockIdx.x;
  const int t = threadIdx.x;
  float s = 0.0f, s2 = 0.0f;
  for (int i = t; i < NNODES; i += 256) {
    const float v = X[(size_t)i * HIDDEN + j];
    s += v; s2 += v * v;
  }
  sh[t] = s; sh2[t] = s2;
  __syncthreads();
  for (int off = 128; off > 0; off >>= 1) {
    if (t < off) { sh[t] += sh[t + off]; sh2[t] += sh2[t + off]; }
    __syncthreads();
  }
  if (t == 0) {
    const float m   = sh[0] * (1.0f / NNODES);
    const float var = sh2[0] * (1.0f / NNODES) - m * m;
    mean[j] = m;
    rstd[j] = rsqrtf(var + EPS);
  }
}

__global__ __launch_bounds__(256)
void bn_apply_kernel(const float* __restrict__ X, const float* __restrict__ mean,
                     const float* __restrict__ rstd, const float* __restrict__ g,
                     const float* __restrict__ b, float* __restrict__ out)
{
  const size_t i = (size_t)blockIdx.x * 256 + threadIdx.x;
  const int j = (int)(i & (HIDDEN - 1));
  out[i] = (X[i] - mean[j]) * rstd[j] * g[j] + b[j];
}

// ---------------------------------------------------------------------------
extern "C" void kernel_launch(void* const* d_in, const int* in_sizes, int n_in,
                              void* d_out, int out_size, void* d_ws, size_t ws_size,
                              hipStream_t stream)
{
  (void)in_sizes; (void)n_in; (void)out_size; (void)ws_size;
  const float* A   = (const float*)d_in[0];
  const float* h   = (const float*)d_in[1];
  const float* Wq  = (const float*)d_in[2];
  const float* bq  = (const float*)d_in[3];
  const float* Wk  = (const float*)d_in[4];
  const float* bk  = (const float*)d_in[5];
  const float* Wv  = (const float*)d_in[6];
  const float* bv  = (const float*)d_in[7];
  const float* Wo  = (const float*)d_in[8];
  const float* bo  = (const float*)d_in[9];
  const float* W1  = (const float*)d_in[10];
  const float* c1  = (const float*)d_in[11];
  const float* W2  = (const float*)d_in[12];
  const float* c2  = (const float*)d_in[13];
  const float* g1  = (const float*)d_in[14];
  const float* be1 = (const float*)d_in[15];
  const float* g2  = (const float*)d_in[16];
  const float* be2 = (const float*)d_in[17];
  float* out = (float*)d_out;

  char* ws = (char*)d_ws;
  const size_t QKV = (size_t)HEADS * NNODES * HEAD_DIM;  // 1,048,576 elems == 4096*256
  __bf16* Qb = (__bf16*)ws;
  __bf16* Kb = Qb + QKV;
  __bf16* Vb = Kb + QKV;
  float* attnF = (float*)(ws + 3 * QKV * sizeof(__bf16));
  float* x0 = attnF + QKV;
  float* x1 = x0 + QKV;
  float* y1 = x1 + QKV;            // [4096, 512]
  float* x2 = y1 + 2 * QKV;
  float* mean1 = x2 + QKV;
  float* rstd1 = mean1 + HIDDEN;
  float* mean2 = rstd1 + HIDDEN;
  float* rstd2 = mean2 + HIDDEN;

  const float qscale = 0.17677669529663687f;  // HEAD_DIM^-0.5
  const dim3 g256(NNODES / 128, HIDDEN / 32);
  const dim3 g512(NNODES / 128, (2 * HIDDEN) / 32);

  // Q/K/V projections (bf16 out for WMMA attention); Q pre-scaled
  gemm_bias_kernel<false, true><<<g256, 256, 0, stream>>>(h, Wq, bq, nullptr, nullptr, Qb,
                                                          NNODES, HIDDEN, HIDDEN, qscale);
  gemm_bias_kernel<false, true><<<g256, 256, 0, stream>>>(h, Wk, bk, nullptr, nullptr, Kb,
                                                          NNODES, HIDDEN, HIDDEN, 1.0f);
  gemm_bias_kernel<false, true><<<g256, 256, 0, stream>>>(h, Wv, bv, nullptr, nullptr, Vb,
                                                          NNODES, HIDDEN, HIDDEN, 1.0f);
  // Fused masked attention
  attn_kernel<<<dim3(HEADS * (NNODES / 64)), 128, 0, stream>>>(Qb, Kb, Vb, A, attnF);
  // Output projection + residual
  gemm_bias_kernel<false, false><<<g256, 256, 0, stream>>>(attnF, Wo, bo, h, x0, nullptr,
                                                           NNODES, HIDDEN, HIDDEN, 1.0f);
  // BatchNorm 1
  bn_stats_kernel<<<HIDDEN, 256, 0, stream>>>(x0, mean1, rstd1);
  bn_apply_kernel<<<(NNODES * HIDDEN) / 256, 256, 0, stream>>>(x0, mean1, rstd1, g1, be1, x1);
  // FFN
  gemm_bias_kernel<true, false><<<g512, 256, 0, stream>>>(x1, W1, c1, nullptr, y1, nullptr,
                                                          NNODES, 2 * HIDDEN, HIDDEN, 1.0f);
  gemm_bias_kernel<false, false><<<g256, 256, 0, stream>>>(y1, W2, c2, x1, x2, nullptr,
                                                           NNODES, HIDDEN, 2 * HIDDEN, 1.0f);
  // BatchNorm 2 -> final output
  bn_stats_kernel<<<HIDDEN, 256, 0, stream>>>(x2, mean2, rstd2);
  bn_apply_kernel<<<(NNODES * HIDDEN) / 256, 256, 0, stream>>>(x2, mean2, rstd2, g2, be2, out);
}
